// SimpleCLIPModel_23029614641664
// MI455X (gfx1250) — compile-verified, matrix operand order checked by reading, not run on hardware
//
#include <hip/hip_runtime.h>
#include <hip/hip_bf16.h>

// ---------------------------------------------------------------------------
// MI455X (gfx1250) implementation.
//  - Fold segment-mean into text side: agg[208(pad),768] bf16, scale 10/count.
//  - GEMM1: Y = X @ W.T + b  (bf16 WMMA, f32 accum), store Yu bf16 + row sumsq.
//    Triple-buffered LDS tiles; async global->LDS DMA overlaps WMMA compute.
//  - GEMM2: out = (Yu * rsqrt(ss)) @ agg.T, -100 for empty classes.
// ---------------------------------------------------------------------------

typedef __bf16 bf16_t;
typedef bf16_t v16bf __attribute__((ext_vector_type(16)));
typedef bf16_t v8bf  __attribute__((ext_vector_type(8)));
typedef bf16_t v4bf  __attribute__((ext_vector_type(4)));
typedef float  v8f   __attribute__((ext_vector_type(8)));
typedef float  v4f   __attribute__((ext_vector_type(4)));
typedef int    v4i   __attribute__((ext_vector_type(4)));

#define D_DIM    768
#define B_ROWS   16384
#define NDEF     2000
#define NCLS     200
#define NCLS_PAD 208
#define BK       32
#define NKT      (D_DIM / BK)   // 24 K-tiles
#define ASTR     40   // LDS row stride (bf16 elems): 80B, 16B-aligned, conflict-free

#define AS1 __attribute__((address_space(1)))
#define AS3 __attribute__((address_space(3)))

#if __has_builtin(__builtin_amdgcn_global_load_async_to_lds_b128)
#define HAVE_ASYNC_LDS 1
#else
#define HAVE_ASYNC_LDS 0
#endif

// 16-byte global->LDS copy; async (no VGPR round-trip, ASYNCcnt) when available.
__device__ __forceinline__ void cp16(bf16_t* lds, const bf16_t* g) {
#if HAVE_ASYNC_LDS
  __builtin_amdgcn_global_load_async_to_lds_b128(
      (AS1 v4i*)(uintptr_t)g, (AS3 v4i*)(uintptr_t)lds, 0, 0);
#else
  *(v8bf*)lds = *(const v8bf*)g;
#endif
}

// Wait until at most N async ops remain outstanding (in-order completion).
template <int N>
__device__ __forceinline__ void wait_async_n() {
#if HAVE_ASYNC_LDS
#if __has_builtin(__builtin_amdgcn_s_wait_asynccnt)
  __builtin_amdgcn_s_wait_asynccnt(N);
#else
  asm volatile("s_wait_asynccnt %0" ::"i"(N) : "memory");
#endif
#endif
}

// Per-lane WMMA 16x16x32 bf16 A/B fragment load:
//   lane<16: K=0..7 then 16..23 ; lane>=16: K=8..15 then 24..31 (ISA 7.12.2)
__device__ __forceinline__ v16bf load_frag(const bf16_t* p) {
  union { v16bf v; v8bf h[2]; } u;
  u.h[0] = *(const v8bf*)(p);
  u.h[1] = *(const v8bf*)(p + 16);
  return u.v;
}

__global__ void k_zero(float* __restrict__ p, int n) {
  int i = blockIdx.x * blockDim.x + threadIdx.x;
  if (i < n) p[i] = 0.0f;
}

// vectorized f32 -> bf16 (n multiple of 4)
__global__ void k_cvt4(const float* __restrict__ s, bf16_t* __restrict__ d, int n4) {
  int i = blockIdx.x * blockDim.x + threadIdx.x;
  if (i < n4) {
    v4f f = *(const v4f*)(s + 4 * (size_t)i);
    v4bf h;
#pragma unroll
    for (int j = 0; j < 4; ++j) h[j] = (bf16_t)f[j];
    *(v4bf*)(d + 4 * (size_t)i) = h;
  }
}

// inverse L2 norm per text row
__global__ __launch_bounds__(256) void k_txtnorm(const float* __restrict__ txt,
                                                 float* __restrict__ invt) {
  __shared__ float red[256];
  const int row = blockIdx.x;
  const int tid = threadIdx.x;
  float s = 0.0f;
  for (int k = tid; k < D_DIM; k += 256) {
    float v = txt[(size_t)row * D_DIM + k];
    s += v * v;
  }
  red[tid] = s;
  __syncthreads();
  for (int off = 128; off > 0; off >>= 1) {
    if (tid < off) red[tid] += red[tid + off];
    __syncthreads();
  }
  if (tid == 0) invt[row] = rsqrtf(fmaxf(red[0], 1e-24f));
}

// per-class scale = 10/count (0 => empty class)
__global__ __launch_bounds__(256) void k_counts(const int* __restrict__ seg,
                                                float* __restrict__ cls_scale) {
  __shared__ int cnt[NCLS];
  const int tid = threadIdx.x;
  if (tid < NCLS) cnt[tid] = 0;
  __syncthreads();
  for (int d = tid; d < NDEF; d += 256) {
    int c = seg[d];
    if (c >= 0 && c < NCLS) atomicAdd(&cnt[c], 1);
  }
  __syncthreads();
  if (tid < NCLS) cls_scale[tid] = cnt[tid] > 0 ? 10.0f / (float)cnt[tid] : 0.0f;
}

// agg[c,k] = (10/count[c]) * sum_{seg[d]==c} txt[d,k]*invnorm[d]; rows 200..207 = 0
__global__ __launch_bounds__(256) void k_agg(const float* __restrict__ txt,
                                             const int* __restrict__ seg,
                                             const float* __restrict__ invt,
                                             const float* __restrict__ cls_scale,
                                             bf16_t* __restrict__ aggb) {
  const int c = blockIdx.x;
  const int tid = threadIdx.x;
  float a0 = 0.0f, a1 = 0.0f, a2 = 0.0f;
  if (c < NCLS) {
    for (int d = 0; d < NDEF; ++d) {
      if (seg[d] == c) {
        float w = invt[d];
        const float* t = txt + (size_t)d * D_DIM;
        a0 += t[tid] * w;
        a1 += t[tid + 256] * w;
        a2 += t[tid + 512] * w;
      }
    }
    float s = cls_scale[c];
    a0 *= s; a1 *= s; a2 *= s;
  }
  bf16_t* o = aggb + (size_t)c * D_DIM;
  o[tid]       = (bf16_t)a0;
  o[tid + 256] = (bf16_t)a1;
  o[tid + 512] = (bf16_t)a2;
}

// GEMM1: tile M128 x N128, K-step 32, triple-buffered LDS (async DMA pipeline).
// 8 waves, each wave = 16 rows x 128 cols.
// USE_XB=true: A tile copied async from pre-converted bf16 X; else convert f32.
template <bool USE_XB>
__global__ __launch_bounds__(256) void k_gemm1(const float* __restrict__ X,
                                               const bf16_t* __restrict__ Xb,
                                               const bf16_t* __restrict__ Wb,
                                               const float* __restrict__ bias,
                                               bf16_t* __restrict__ Yu,
                                               float* __restrict__ rowSS) {
  __shared__ __align__(16) bf16_t ldsA[3][128 * ASTR];
  __shared__ __align__(16) bf16_t ldsB[3][128 * ASTR];
  const int tid   = threadIdx.x;
  const int wave  = tid >> 5;
  const int lane  = tid & 31;
  const int nl    = lane & 15;
  const int khalf = (lane >> 4) << 3;   // 0 or 8
  const int m0 = blockIdx.x * 128;
  const int n0 = blockIdx.y * 128;

  const v8f vz = {0.f, 0.f, 0.f, 0.f, 0.f, 0.f, 0.f, 0.f};
  v8f acc[8];
#pragma unroll
  for (int i = 0; i < 8; ++i) acc[i] = vz;

  const int srow = tid >> 1;        // staging row 0..127
  const int sks  = (tid & 1) << 4;  // staging k offset 0 or 16
  const int soff = srow * ASTR + sks;
  const bf16_t* gB = Wb + (size_t)(n0 + srow) * D_DIM + sks;
  const bf16_t* gA = Xb + (size_t)(m0 + srow) * D_DIM + sks;
  const float*  gX = X  + (size_t)(m0 + srow) * D_DIM + sks;

  // issue staging (4 async b128 ops per thread per tile)
  auto stage = [&](int t) {
    const int buf = t % 3;
    const int kk  = t * BK;
    if constexpr (USE_XB) {
      cp16(ldsA[buf] + soff, gA + kk);
      cp16(ldsA[buf] + soff + 8, gA + kk + 8);
    } else {
      v4f f0 = *(const v4f*)(gX + kk);
      v4f f1 = *(const v4f*)(gX + kk + 4);
      v4f f2 = *(const v4f*)(gX + kk + 8);
      v4f f3 = *(const v4f*)(gX + kk + 12);
      v8bf h0, h1;
#pragma unroll
      for (int i = 0; i < 4; ++i) {
        h0[i] = (bf16_t)f0[i]; h0[4 + i] = (bf16_t)f1[i];
        h1[i] = (bf16_t)f2[i]; h1[4 + i] = (bf16_t)f3[i];
      }
      *(v8bf*)(ldsA[buf] + soff)     = h0;
      *(v8bf*)(ldsA[buf] + soff + 8) = h1;
    }
    cp16(ldsB[buf] + soff, gB + kk);
    cp16(ldsB[buf] + soff + 8, gB + kk + 8);
  };

  stage(0);
  stage(1);
  for (int i = 0; i < NKT; ++i) {
    // tile i done when <= 4 ops (tile i+1) remain outstanding (in-order)
    if (i < NKT - 1) wait_async_n<4>();
    else             wait_async_n<0>();
    __syncthreads();
    // prefetch tile i+2 into the buffer last read at tile i-1 (safe post-barrier)
    if (i + 2 < NKT) stage(i + 2);

    const bf16_t* bA = ldsA[i % 3];
    const bf16_t* bB = ldsB[i % 3];
    v16bf a = load_frag(bA + (wave * 16 + nl) * ASTR + khalf);
    v16bf bfr[8];
#pragma unroll
    for (int nt = 0; nt < 8; ++nt)
      bfr[nt] = load_frag(bB + (nt * 16 + nl) * ASTR + khalf);
#pragma unroll
    for (int nt = 0; nt < 8; ++nt)
      acc[nt] = __builtin_amdgcn_wmma_f32_16x16x32_bf16(
          false, a, false, bfr[nt], (short)0, acc[nt], false, false);
  }

  // epilogue: + bias, partial row sum-of-squares, store bf16
  float rsq[8] = {0, 0, 0, 0, 0, 0, 0, 0};
#pragma unroll
  for (int nt = 0; nt < 8; ++nt) {
    float bv = bias[n0 + nt * 16 + nl];
#pragma unroll
    for (int r = 0; r < 8; ++r) {
      float v = acc[nt][r] + bv;
      acc[nt][r] = v;
      rsq[r] += v * v;
    }
  }
  // reduce over the 16 lanes sharing each row (xor<16 stays within half-wave)
#pragma unroll
  for (int m = 1; m < 16; m <<= 1)
#pragma unroll
    for (int r = 0; r < 8; ++r) rsq[r] += __shfl_xor(rsq[r], m, 32);

  const int rbase = m0 + wave * 16 + ((lane >> 4) << 3);
  if (nl == 0) {
#pragma unroll
    for (int r = 0; r < 8; ++r) atomicAdd(&rowSS[rbase + r], rsq[r]);
  }
#pragma unroll
  for (int nt = 0; nt < 8; ++nt) {
    int ncol = n0 + nt * 16 + nl;
#pragma unroll
    for (int r = 0; r < 8; ++r)
      Yu[(size_t)(rbase + r) * D_DIM + ncol] = (bf16_t)acc[nt][r];
  }
}

// GEMM2: tile M64 x N208 (13 tiles). 4 waves, each = 16 rows x 208 cols.
// A staged with fused normalization; B (agg) staged async into LDS per K-step.
__global__ __launch_bounds__(128) void k_gemm2(const bf16_t* __restrict__ Yu,
                                               const bf16_t* __restrict__ Aggb,
                                               const float* __restrict__ rowSS,
                                               const float* __restrict__ cls_scale,
                                               float* __restrict__ out) {
  __shared__ __align__(16) bf16_t ldsA[64 * ASTR];
  __shared__ __align__(16) bf16_t ldsB[NCLS_PAD * ASTR];
  __shared__ float invn[64];
  const int tid   = threadIdx.x;
  const int wave  = tid >> 5;
  const int lane  = tid & 31;
  const int nl    = lane & 15;
  const int khalf = (lane >> 4) << 3;
  const int m0 = blockIdx.x * 64;

  if (tid < 64) invn[tid] = rsqrtf(fmaxf(rowSS[m0 + tid], 1e-24f));
  __syncthreads();

  const v8f vz = {0.f, 0.f, 0.f, 0.f, 0.f, 0.f, 0.f, 0.f};
  v8f acc[13];
#pragma unroll
  for (int i = 0; i < 13; ++i) acc[i] = vz;

  const int srow = tid >> 1;        // 0..63
  const int sks  = (tid & 1) << 4;  // 0 or 16
  const bf16_t* gA = Yu + (size_t)(m0 + srow) * D_DIM + sks;
  bf16_t* lA = ldsA + srow * ASTR + sks;
  const float ascale = invn[srow];

  for (int kk = 0; kk < D_DIM; kk += BK) {
    // stage B tile: 208 rows x 16 k (k 0..7, 16..23), pure bf16 copy -> async (416 16B chunks)
#pragma unroll
    for (int j = 0; j < 4; ++j) {
      int c = tid + 128 * j;
      if (c < 2 * NCLS_PAD) {
        int row = c >> 1;
        int ks2 = (c & 1) << 4;
        cp16(ldsB + row * ASTR + ks2,
             Aggb + (size_t)row * D_DIM + kk + ks2);
      }
    }
    // stage A tile with fused row normalization
    {
      v8bf h0 = *(const v8bf*)(gA + kk);
      v8bf h1 = *(const v8bf*)(gA + kk + 8);
      v8bf o0, o1;
#pragma unroll
      for (int i = 0; i < 8; ++i) {
        o0[i] = (bf16_t)((float)h0[i] * ascale);
        o1[i] = (bf16_t)((float)h1[i] * ascale);
      }
      *(v8bf*)(lA)     = o0;
      *(v8bf*)(lA + 8) = o1;
    }
    wait_async_n<0>();
    __syncthreads();

    v16bf a = load_frag(ldsA + (wave * 16 + nl) * ASTR + khalf);
    // chunks of 4 B-fragments so DS waits batch instead of per-WMMA drains
#pragma unroll
    for (int base = 0; base < 12; base += 4) {
      v16bf bfr[4];
#pragma unroll
      for (int j = 0; j < 4; ++j)
        bfr[j] = load_frag(ldsB + ((base + j) * 16 + nl) * ASTR + khalf);
#pragma unroll
      for (int j = 0; j < 4; ++j)
        acc[base + j] = __builtin_amdgcn_wmma_f32_16x16x32_bf16(
            false, a, false, bfr[j], (short)0, acc[base + j], false, false);
    }
    {
      v16bf b12 = load_frag(ldsB + (12 * 16 + nl) * ASTR + khalf);
      acc[12] = __builtin_amdgcn_wmma_f32_16x16x32_bf16(
          false, a, false, b12, (short)0, acc[12], false, false);
    }
    __syncthreads();
  }

  const int rbase = m0 + wave * 16 + ((lane >> 4) << 3);
#pragma unroll
  for (int nt = 0; nt < 13; ++nt) {
    int n = nt * 16 + nl;
    if (n < NCLS) {
      bool live = cls_scale[n] > 0.0f;
#pragma unroll
      for (int r = 0; r < 8; ++r)
        out[(size_t)(rbase + r) * NCLS + n] = live ? acc[nt][r] : -100.0f;
    }
  }
}

extern "C" void kernel_launch(void* const* d_in, const int* in_sizes, int n_in,
                              void* d_out, int out_size, void* d_ws, size_t ws_size,
                              hipStream_t stream) {
  const float* X    = (const float*)d_in[0];   // [16384,768]
  const float* W    = (const float*)d_in[1];   // [768,768]
  const float* bias = (const float*)d_in[2];   // [768]
  const float* txt  = (const float*)d_in[3];   // [2000,768]
  const int*   seg  = (const int*)d_in[4];     // [2000]
  float* out = (float*)d_out;                  // [16384,200]

  char* ws = (char*)d_ws;
  size_t off = 0;
  auto alloc = [&](size_t bytes) -> void* {
    void* p = ws + off;
    off += (bytes + 255) & ~(size_t)255;
    return p;
  };
  bf16_t* Wb     = (bf16_t*)alloc((size_t)D_DIM * D_DIM * 2);
  bf16_t* Yu     = (bf16_t*)alloc((size_t)B_ROWS * D_DIM * 2);
  bf16_t* Aggb   = (bf16_t*)alloc((size_t)NCLS_PAD * D_DIM * 2);
  float*  rowSS  = (float*)alloc((size_t)B_ROWS * 4);
  float*  invt   = (float*)alloc((size_t)NDEF * 4);
  float*  cscale = (float*)alloc(256 * 4);
  bf16_t* Xb     = (bf16_t*)alloc((size_t)B_ROWS * D_DIM * 2);
  const bool use_xb = (off <= ws_size);

  k_zero<<<(B_ROWS + 255) / 256, 256, 0, stream>>>(rowSS, B_ROWS);
  k_cvt4<<<(D_DIM * D_DIM / 4 + 255) / 256, 256, 0, stream>>>(W, Wb, D_DIM * D_DIM / 4);
  k_txtnorm<<<NDEF, 256, 0, stream>>>(txt, invt);
  k_counts<<<1, 256, 0, stream>>>(seg, cscale);
  k_agg<<<NCLS_PAD, 256, 0, stream>>>(txt, seg, invt, cscale, Aggb);

  dim3 g1(B_ROWS / 128, D_DIM / 128);
  if (use_xb) {
    k_cvt4<<<((B_ROWS * D_DIM / 4) + 255) / 256, 256, 0, stream>>>(
        X, Xb, B_ROWS * D_DIM / 4);
    k_gemm1<true><<<g1, 256, 0, stream>>>(X, Xb, Wb, bias, Yu, rowSS);
  } else {
    k_gemm1<false><<<g1, 256, 0, stream>>>(X, Xb, Wb, bias, Yu, rowSS);
  }
  k_gemm2<<<B_ROWS / 64, 128, 0, stream>>>(Yu, Aggb, rowSS, cscale, out);
}
